// LinearVQVAE_22539988370207
// MI455X (gfx1250) — compile-verified, matrix operand order checked by reading, not run on hardware
//
#include <hip/hip_runtime.h>
#include <stdint.h>

// ---------------- types ----------------
typedef __attribute__((ext_vector_type(16))) __bf16          v16bf;
typedef __attribute__((ext_vector_type(8)))  float           v8f;
typedef __attribute__((ext_vector_type(4)))  float           v4f;
typedef __attribute__((ext_vector_type(8)))  unsigned short  v8u;
typedef __attribute__((ext_vector_type(16))) unsigned short  v16u;
typedef __attribute__((ext_vector_type(8)))  unsigned int    v8ui;
typedef __attribute__((ext_vector_type(4)))  unsigned int    v4ui;

#define TOKENS 65536
#define DIN    1024
#define DLAT   256
#define NCODES 128
#define ZPITCH (DLAT + 8)   // +16B/row: staggers banks for the strided frag reads

// ---------------- helpers ----------------
// Packed fp32->bf16 (RNE): one VALU op converts+packs two elements.
__device__ __forceinline__ unsigned int cvt_pk_bf16(float a, float b) {
  unsigned int r;
  asm("v_cvt_pk_bf16_f32 %0, %1, %2" : "=v"(r) : "v"(a), "v"(b));
  return r;
}

__device__ __forceinline__ unsigned short f2bf(float f) {  // scalar (prep only)
  unsigned int x = __builtin_bit_cast(unsigned int, f);
  x += 0x7fffu + ((x >> 16) & 1u);
  return (unsigned short)(x >> 16);
}

__device__ __forceinline__ v16bf pack_frag(v4f a0, v4f a1, v4f b0, v4f b1) {
  v8ui w;
  w[0] = cvt_pk_bf16(a0[0], a0[1]); w[1] = cvt_pk_bf16(a0[2], a0[3]);
  w[2] = cvt_pk_bf16(a1[0], a1[1]); w[3] = cvt_pk_bf16(a1[2], a1[3]);
  w[4] = cvt_pk_bf16(b0[0], b0[1]); w[5] = cvt_pk_bf16(b0[2], b0[3]);
  w[6] = cvt_pk_bf16(b1[0], b1[1]); w[7] = cvt_pk_bf16(b1[2], b1[3]);
  return __builtin_bit_cast(v16bf, w);
}

// fragment from fp32 row (cached path; dec_cb kernel operands)
__device__ __forceinline__ v16bf load_frag_f32(const float* row, int koff) {
  const v4f* p0 = (const v4f*)(row + koff);
  const v4f* p1 = (const v4f*)(row + koff + 16);
  return pack_frag(p0[0], p0[1], p1[0], p1[1]);
}

// fragment from fp32 row, non-temporal (x is streamed exactly once)
__device__ __forceinline__ v16bf load_frag_f32_nt(const float* row, int koff) {
  const v4f* p0 = (const v4f*)(row + koff);
  const v4f* p1 = (const v4f*)(row + koff + 16);
  v4f a0 = __builtin_nontemporal_load(p0);
  v4f a1 = __builtin_nontemporal_load(p0 + 1);
  v4f b0 = __builtin_nontemporal_load(p1);
  v4f b1 = __builtin_nontemporal_load(p1 + 1);
  return pack_frag(a0, a1, b0, b1);
}

// fragment from a bf16 row: two 16B chunks at +0 / +16 halves, register concat
__device__ __forceinline__ v16bf load_frag_bf16(const unsigned short* row, int koff) {
  v8u lo = *(const v8u*)(row + koff);
  v8u hi = *(const v8u*)(row + koff + 16);
  v16u t = __builtin_shufflevector(lo, hi, 0, 1, 2, 3, 4, 5, 6, 7,
                                   8, 9, 10, 11, 12, 13, 14, 15);
  return __builtin_bit_cast(v16bf, t);
}

__device__ __forceinline__ v8f wmma_bf16(v16bf a, v16bf b, v8f c) {
  return __builtin_amdgcn_wmma_f32_16x16x32_bf16(false, a, false, b, (short)0, c,
                                                 false, false);
}

// ---------------- kernel 1: blocked bf16 weights, permuted codebook, norms ----
// wblk[k][n][32]  : W_enc k-blocked -> each 16x32 B-tile read is a dense 1KB
// cbblk[k][c][32] : codebook k-blocked over the *permuted* latent axis
//                   (perm(l) = ((l&15)<<4)|(l>>4), an involution; distances are
//                   invariant under a shared permutation of the contraction axis)
__global__ void prep_kernel(const float* __restrict__ Wenc,
                            const float* __restrict__ cb,
                            unsigned short* __restrict__ wblk,
                            unsigned short* __restrict__ cbblk,
                            float* __restrict__ e_sq) {
  int tid = blockIdx.x * blockDim.x + threadIdx.x;
  int nt  = gridDim.x * blockDim.x;
  for (int i = tid; i < DLAT * DIN; i += nt) {
    int n = i >> 10, d = i & 1023;
    int k = d >> 5, j = d & 31;
    wblk[((size_t)k * DLAT + n) * 32 + j] = f2bf(Wenc[i]);
  }
  for (int i = tid; i < NCODES * DLAT; i += nt) {
    int c = i >> 8, lp = i & 255;                    // lp = permuted latent idx
    int l = ((lp & 15) << 4) | (lp >> 4);            // source (unpermuted) idx
    int k = lp >> 5, j = lp & 31;
    cbblk[((size_t)k * NCODES + c) * 32 + j] = f2bf(cb[c * DLAT + l]);
  }
  for (int r = tid; r < NCODES; r += nt) {
    float s = 0.f;
    const float* row = cb + (size_t)r * DLAT;
    for (int j = 0; j < DLAT; ++j) s += row[j] * row[j];
    e_sq[r] = s;
  }
}

// ---------------- kernel 2: dec_cb = codebook @ W_dec^T + b_dec --------------
__global__ __launch_bounds__(32) void dec_cb_kernel(
    const float* __restrict__ cb, const float* __restrict__ Wdec,
    const float* __restrict__ bdec, float* __restrict__ dec_cb) {
  int mt = blockIdx.x >> 6;      // 0..7
  int nt = blockIdx.x & 63;      // 0..63
  int lane = threadIdx.x & 31;
  int kh   = lane >> 4;
  int col  = lane & 15;
  v8f acc = {};
  const float* arow = cb   + (size_t)(mt * 16 + col) * DLAT;
  const float* brow = Wdec + (size_t)(nt * 16 + col) * DLAT;
#pragma unroll
  for (int k = 0; k < 8; ++k) {
    int koff = k * 32 + kh * 8;
    acc = wmma_bf16(load_frag_f32(arow, koff), load_frag_f32(brow, koff), acc);
  }
  float bias = bdec[nt * 16 + col];
#pragma unroll
  for (int v = 0; v < 8; ++v)
    dec_cb[(size_t)(mt * 16 + v + 8 * kh) * DIN + nt * 16 + col] = acc[v] + bias;
}

// ---------------- kernel 3: fused encode + VQ + gather-decode ----------------
__global__ __launch_bounds__(128) void vq_main_kernel(
    const float* __restrict__ x, const float* __restrict__ benc,
    const unsigned short* __restrict__ wblk,    // W_enc, k-blocked bf16
    const unsigned short* __restrict__ cbblk,   // codebook, k-blocked, permuted
    const float* __restrict__ e_sq, const float* __restrict__ dec_cb,
    float* __restrict__ out, int* __restrict__ idx_out,
    float* __restrict__ partials) {
  __shared__ __align__(16) unsigned short z_lds[64][ZPITCH];  // z, bf16, permuted
  __shared__ float d2_lds[64];
  __shared__ int   idx_lds[64];

  const int wave = threadIdx.x >> 5;
  const int lane = threadIdx.x & 31;
  const int kh   = lane >> 4;
  const int col  = lane & 15;
  const int tok_base = blockIdx.x * 64 + wave * 16;

  // ---- Phase 1: z = x @ W_enc^T (16 live acc tiles, k-outer, A reused 16x)
  v8f acc[16] = {};
  const float* xrow = x + (size_t)(tok_base + col) * DIN;
  const unsigned short* wlane = wblk + col * 32 + kh * 8;  // per-lane B base
  for (int k = 0; k < 32; ++k) {
    int koff = k * 32 + kh * 8;
    __builtin_prefetch(xrow + koff + 32, 0, 0);
    v16bf a = load_frag_f32_nt(xrow, koff);
    const unsigned short* wk = wlane + (size_t)k * (DLAT * 32);  // 16KB slice
#pragma unroll
    for (int n = 0; n < 16; ++n) {
      // one 16x32 B-tile = dense 1KB block; chunks at +0 / +16 halves
      v16bf b = load_frag_bf16(wk + n * 512, 0);
      acc[n] = wmma_bf16(a, b, acc[n]);
    }
  }

  // ---- bias add + |z|^2 + packed bf16 store to LDS (permuted layout)
  // C/D layout: lane holds lat col = n*16+col, token row = v + 8*kh.
  // Permuted store: token row, halfword position col*16 + n  (contiguous!).
  float bias[16];
#pragma unroll
  for (int n = 0; n < 16; ++n) bias[n] = benc[n * 16 + col];

  float zsq[8];
#pragma unroll
  for (int v = 0; v < 8; ++v) {
    v4ui w0, w1;
    float s = 0.f;
#pragma unroll
    for (int n2 = 0; n2 < 4; ++n2) {
      float z0 = acc[2 * n2][v] + bias[2 * n2];
      float z1 = acc[2 * n2 + 1][v] + bias[2 * n2 + 1];
      s += z0 * z0 + z1 * z1;
      w0[n2] = cvt_pk_bf16(z0, z1);
    }
#pragma unroll
    for (int n2 = 4; n2 < 8; ++n2) {
      float z0 = acc[2 * n2][v] + bias[2 * n2];
      float z1 = acc[2 * n2 + 1][v] + bias[2 * n2 + 1];
      s += z0 * z0 + z1 * z1;
      w1[n2 - 4] = cvt_pk_bf16(z0, z1);
    }
    unsigned short* rp = &z_lds[wave * 16 + v + 8 * kh][col * 16];
    *(v4ui*)(rp)     = w0;   // ds_store_b128
    *(v4ui*)(rp + 8) = w1;   // ds_store_b128
    zsq[v] = s;
  }
  // sum |z|^2 across the 16 column lanes of each half
#pragma unroll
  for (int off = 1; off < 16; off <<= 1)
#pragma unroll
    for (int v = 0; v < 8; ++v) zsq[v] += __shfl_xor(zsq[v], off, 32);

  // ---- Phase 2: scores = e_sq - 2*(z @ cb^T) over permuted latent axis.
  // Same-wave DS ordering makes the z stores visible to these loads.
  v16bf za[8];
  const unsigned short* zrow = z_lds[wave * 16 + col];
#pragma unroll
  for (int k = 0; k < 8; ++k) za[k] = load_frag_bf16(zrow, k * 32 + kh * 8);

  float minv[8]; int mini[8];
#pragma unroll
  for (int v = 0; v < 8; ++v) { minv[v] = 3.0e38f; mini[v] = 0; }

  const unsigned short* clane = cbblk + col * 32 + kh * 8;   // per-lane B base
  for (int ct = 0; ct < 8; ++ct) {
    v8f s = {};
#pragma unroll
    for (int k = 0; k < 8; ++k)
      s = wmma_bf16(za[k], load_frag_bf16(clane + k * (NCODES * 32) + ct * 512, 0), s);
    int code = ct * 16 + col;
    float es = e_sq[code];
#pragma unroll
    for (int v = 0; v < 8; ++v) {
      float d = es - 2.0f * s[v];
      if (d < minv[v]) { minv[v] = d; mini[v] = code; }
    }
  }
  // argmin across the 16 column lanes (tie-break: lowest index, like jnp.argmin)
#pragma unroll
  for (int off = 1; off < 16; off <<= 1) {
#pragma unroll
    for (int v = 0; v < 8; ++v) {
      float ov = __shfl_xor(minv[v], off, 32);
      int   oi = __shfl_xor(mini[v], off, 32);
      if (ov < minv[v] || (ov == minv[v] && oi < mini[v])) {
        minv[v] = ov; mini[v] = oi;
      }
    }
  }
  if (col == 0) {   // lanes 0 (tokens 0..7) and 16 (tokens 8..15)
#pragma unroll
    for (int v = 0; v < 8; ++v) {
      int tl = v + 8 * kh;
      int tg = tok_base + tl;
      idx_out[tg] = mini[v];
      idx_lds[wave * 16 + tl] = mini[v];
      d2_lds[wave * 16 + tl] = zsq[v] + minv[v];   // = d2_min for commit loss
    }
  }
  __syncthreads();

  if (threadIdx.x == 0) {   // deterministic fixed-order per-block partial
    float s = 0.f;
    for (int i = 0; i < 64; ++i) s += d2_lds[i];
    partials[blockIdx.x] = s;
  }

  // ---- Phase 3: out[t,:] = dec_cb[idx[t],:]  (gather; NT streaming stores)
  for (int t = 0; t < 16; ++t) {
    int idx = idx_lds[wave * 16 + t];     // same-wave LDS, in-order
    const v4f* src = (const v4f*)(dec_cb + (size_t)idx * DIN);
    v4f* dst = (v4f*)(out + (size_t)(tok_base + t) * DIN);
#pragma unroll
    for (int i = 0; i < 8; ++i)
      __builtin_nontemporal_store(src[lane + 32 * i], dst + lane + 32 * i);
  }
}

// ---------------- kernel 4: deterministic loss finalize ----------------
__global__ void finalize_kernel(const float* __restrict__ partials,
                                float* __restrict__ loss) {
  if (threadIdx.x == 0 && blockIdx.x == 0) {
    float s = 0.f;
    for (int i = 0; i < 1024; ++i) s += partials[i];
    loss[0] = s * (1.0f / ((float)TOKENS * (float)DLAT));
  }
}

// ---------------- launch ----------------
extern "C" void kernel_launch(void* const* d_in, const int* in_sizes, int n_in,
                              void* d_out, int out_size, void* d_ws, size_t ws_size,
                              hipStream_t stream) {
  const float* x    = (const float*)d_in[0];
  const float* Wenc = (const float*)d_in[1];
  const float* benc = (const float*)d_in[2];
  const float* cb   = (const float*)d_in[3];
  const float* Wdec = (const float*)d_in[4];
  const float* bdec = (const float*)d_in[5];

  // workspace layout (~1.1 MB total)
  char* ws = (char*)d_ws;
  unsigned short* wblk     = (unsigned short*)(ws);                  // 512 KB
  unsigned short* cbblk    = (unsigned short*)(ws + 524288);         //  64 KB
  float*          e_sq     = (float*)(ws + 524288 + 65536);          // 512 B
  float*          dec_cb   = (float*)(ws + 524288 + 65536 + 512);    // 512 KB
  float*          partials = (float*)(ws + 1114624);                 //   4 KB

  float* outp  = (float*)d_out;
  int*   idxp  = (int*)(outp + (size_t)TOKENS * DIN);
  float* lossp = (float*)(outp + (size_t)TOKENS * DIN + TOKENS);

  prep_kernel<<<64, 256, 0, stream>>>(Wenc, cb, wblk, cbblk, e_sq);
  dec_cb_kernel<<<512, 32, 0, stream>>>(cb, Wdec, bdec, dec_cb);
  vq_main_kernel<<<1024, 128, 0, stream>>>(x, benc, wblk, cbblk, e_sq, dec_cb,
                                           outp, idxp, partials);
  finalize_kernel<<<1, 32, 0, stream>>>(partials, lossp);
}
